// HeteroFloodGNN_39694087750147
// MI455X (gfx1250) — matrix operand order repaired and checked by reading, not compiled, not run on hardware
//
#include <hip/hip_runtime.h>
#include <math.h>

// ---------------------------------------------------------------------------
// HeteroFloodGNN on MI455X (gfx1250, wave32, WMMA)
//
// Dense projections run through v_wmma_f32_16x16x32_f16 with:
//   - compile-time K / K0 / Npad (templated) so the [x|h] split is branch-free
//   - A tiles loaded as 4x global_load_b128 (aligned float4 runs per lane)
//   - W pre-packed into native B-fragment layout, staged once per block into
//     LDS (ds_load_b128 in the inner loop; 4 waves share one copy)
//   - one wave computes a 16xNpad strip (A fragment reused across N tiles)
// Edge softmax/scatter is fp32 VALU + atomics with float4 feature reads.
// ---------------------------------------------------------------------------

typedef __attribute__((ext_vector_type(16))) _Float16 v16h;
typedef __attribute__((ext_vector_type(8)))  float    v8f;

static constexpr int kNM = 25000;    // manholes
static constexpr int kNC = 100000;   // cells
static constexpr int kEMM = 75000;
static constexpr int kECC = 400000;
static constexpr int kEMC = 50000;
static constexpr int kECM = 50000;

// ---------------------------------------------------------------------------
// WMMA GEMM:  C[M,NP] = act( [A0 | A1][M,K] @ W[K,NP] + bias )
//   A0 row-major width K0 (fp32); A1 row-major width K-K0 supplies cols K0..K-1.
//   K, K0 multiples of 32 -> the A0/A1 choice is per-32-chunk, compile time.
//   Wp: f16 fragments packed as [chunk][ntile][lane][16] (see pack_w16),
//       staged into LDS once per block.
//   Block = 128 threads = 4 waves; each wave owns one 16-row strip.
// ISA 7.12.2 layouts:
//   A 16x32 f16 : lane&15 = row; lanes 0-15 K {0..7,16..23}, 16-31 {8..15,24..31}
//   B 32x16 f16 : lane&15 = col; lanes 0-15 K {0..7,16..23}, 16-31 {8..15,24..31}
//   C  16x16 f32: lane&15 = col; VGPR r -> row r + 8*(lane>>4)
// ---------------------------------------------------------------------------
template <int K, int K0, int NP>
__global__ void gemm_wmma(const float* __restrict__ A0,
                          const float* __restrict__ A1,
                          const _Float16* __restrict__ Wp,
                          const float* __restrict__ bias, int Nreal,
                          float* __restrict__ C, int M, int act)
{
  constexpr int NT = NP / 16;               // number of 16-col tiles
  constexpr int KB = (K > K0) ? (K - K0) : 1;
  constexpr int FRAG = (K / 32) * NT * 512; // f16 elements in weight slab

  // ---- stage packed W into LDS (shared by the 4 waves of this block) ----
  __shared__ __align__(32) _Float16 Bsh[FRAG];
  {
    const v16h* __restrict__ src = (const v16h*)Wp;
    v16h* dst = (v16h*)Bsh;
#pragma unroll
    for (int i = threadIdx.x; i < FRAG / 16; i += 128) dst[i] = src[i];
  }
  __syncthreads();

  const int lane = threadIdx.x & 31;
  const int wave = threadIdx.x >> 5;
  const int idx  = lane & 15;               // row (A) / col (B,C)
  const int half = lane >> 4;
  const int m0 = (blockIdx.x * 4 + wave) * 16;
  const int gm = m0 + idx;
  const bool mv = (gm < M);

  v8f acc[NT] = {};

#pragma unroll
  for (int kc = 0; kc < K; kc += 32) {
    // ---- A fragment: two aligned 8-float runs per lane ----
    float4 f0 = {0.f, 0.f, 0.f, 0.f}, f1 = f0, f2 = f0, f3 = f0;
    if (mv) {
      if (kc < K0) {   // compile-time per unrolled iteration
        const float* __restrict__ R = A0 + (size_t)gm * K0 + kc;
        const int kb = half * 8;
        f0 = *(const float4*)(R + kb);
        f1 = *(const float4*)(R + kb + 4);
        f2 = *(const float4*)(R + kb + 16);
        f3 = *(const float4*)(R + kb + 20);
      } else {
        const float* __restrict__ R = A1 + (size_t)gm * KB + (kc - K0);
        const int kb = half * 8;
        f0 = *(const float4*)(R + kb);
        f1 = *(const float4*)(R + kb + 4);
        f2 = *(const float4*)(R + kb + 16);
        f3 = *(const float4*)(R + kb + 20);
      }
    }
    v16h a;
    a[0] = (_Float16)f0.x; a[1] = (_Float16)f0.y;
    a[2] = (_Float16)f0.z; a[3] = (_Float16)f0.w;
    a[4] = (_Float16)f1.x; a[5] = (_Float16)f1.y;
    a[6] = (_Float16)f1.z; a[7] = (_Float16)f1.w;
    a[8]  = (_Float16)f2.x; a[9]  = (_Float16)f2.y;
    a[10] = (_Float16)f2.z; a[11] = (_Float16)f2.w;
    a[12] = (_Float16)f3.x; a[13] = (_Float16)f3.y;
    a[14] = (_Float16)f3.z; a[15] = (_Float16)f3.w;

    // ---- B fragments from LDS (contiguous 32B per lane) + WMMA ----
#pragma unroll
    for (int nt = 0; nt < NT; ++nt) {
      const _Float16* bp = Bsh + (((kc / 32) * NT + nt) * 32 + lane) * 16;
      v16h b = *(const v16h*)bp;
      acc[nt] = __builtin_amdgcn_wmma_f32_16x16x32_f16(
          /*neg_a=*/false, a, /*neg_b=*/false, b,
          /*c_mod=*/(short)0, acc[nt], /*reuse_a=*/false, /*reuse_b=*/false);
    }
  }

  // ---- bias + activation + store ----
#pragma unroll
  for (int nt = 0; nt < NT; ++nt) {
    const int gn = nt * 16 + idx;
    const float bv = (gn < Nreal) ? bias[gn] : 0.f;
#pragma unroll
    for (int r = 0; r < 8; ++r) {
      const int m = m0 + half * 8 + r;
      if (m < M) {
        float v = acc[nt][r] + bv;
        if (act == 1) v = fmaxf(v, 0.f);   // ReLU
        C[(size_t)m * NP + gn] = v;
      }
    }
  }
}

// ---------------------------------------------------------------------------
// Pack fp32 weight [K,N] into f16 B-fragments, zero-padded to Kpad x Npad.
// dst[t] with t = ((chunk*NT + ntile)*32 + lane)*16 + j
//   k = chunk*32 + (j>>3)*16 + (lane>>4)*8 + (j&7) ; n = ntile*16 + (lane&15)
// ---------------------------------------------------------------------------
__global__ void pack_w16(const float* __restrict__ src,
                         _Float16* __restrict__ dst,
                         int K, int N, int Kpad, int Npad)
{
  const int NT = Npad / 16;
  const int total = (Kpad / 32) * NT * 512;
  const int t = blockIdx.x * blockDim.x + threadIdx.x;
  if (t >= total) return;
  const int j = t & 15;
  const int lane = (t >> 4) & 31;
  const int tile = t >> 9;
  const int nt = tile % NT;
  const int chunk = tile / NT;
  const int k = chunk * 32 + (j >> 3) * 16 + (lane >> 4) * 8 + (j & 7);
  const int n = nt * 16 + (lane & 15);
  const float v = (k < K && n < N) ? src[k * N + n] : 0.f;
  dst[t] = (_Float16)v;
}

// Pack [static | dyn*scale | 0-pad] into a 32-wide encoder input row
__global__ void pack_inputs(const float* __restrict__ stat, int sdim,
                            const float* __restrict__ dyn, int ddim,
                            float* __restrict__ out, int n, float scale)
{
  const int t = blockIdx.x * blockDim.x + threadIdx.x;
  if (t >= n * 32) return;
  const int i = t / 32, c = t % 32;
  float v = 0.f;
  if (c < sdim) v = stat[i * sdim + c];
  else if (c < sdim + ddim) v = dyn[i * ddim + (c - sdim)] * scale;
  out[t] = v;
}

__global__ void fill_f32(float* __restrict__ p, int n, float v)
{
  const int t = blockIdx.x * blockDim.x + threadIdx.x;
  if (t < n) p[t] = v;
}

// raw[nd*64]=0, smax[nd*4]=-inf, sden[nd*4]=0
__global__ void seg_init(float* __restrict__ raw, float* __restrict__ smax,
                         float* __restrict__ sden, int nd)
{
  const int t = blockIdx.x * blockDim.x + threadIdx.x;
  if (t >= nd * 64) return;
  raw[t] = 0.f;
  if (t < nd * 4) { smax[t] = -INFINITY; sden[t] = 0.f; }
}

__device__ inline void atomicMaxF(float* addr, float val)
{
  unsigned int* u = (unsigned int*)addr;
  unsigned int old = *u;
  while (__uint_as_float(old) < val) {
    const unsigned int assumed = old;
    old = atomicCAS(u, assumed, __float_as_uint(val));
    if (old == assumed) break;
  }
}

// Pass 1: logit[e,h] = att_h . leakyrelu(gl[src]+gr[dst]); atomic-max per dst.
// e >= E synthesizes self-loops (src = dst = e - E).
__global__ void edge_logits_max(const int* __restrict__ eidx, int E, int nself,
                                const float* __restrict__ gl,
                                const float* __restrict__ gr,
                                const float* __restrict__ att,  // [4,16]
                                float* __restrict__ elog,
                                float* __restrict__ smax)
{
  const int e = blockIdx.x * blockDim.x + threadIdx.x;
  if (e >= E + nself) return;
  int s, d;
  if (e < E) { s = eidx[e]; d = eidx[E + e]; } else { s = e - E; d = s; }
  const float4* pl = (const float4*)(gl + (size_t)s * 64);
  const float4* pr = (const float4*)(gr + (size_t)d * 64);
  const float4* at = (const float4*)att;
#pragma unroll
  for (int h = 0; h < 4; ++h) {
    float logit = 0.f;
#pragma unroll
    for (int q = 0; q < 4; ++q) {
      const float4 l = pl[h * 4 + q];
      const float4 r = pr[h * 4 + q];
      const float4 w = at[h * 4 + q];
      float v;
      v = l.x + r.x; v = (v > 0.f) ? v : 0.2f * v; logit += v * w.x;
      v = l.y + r.y; v = (v > 0.f) ? v : 0.2f * v; logit += v * w.y;
      v = l.z + r.z; v = (v > 0.f) ? v : 0.2f * v; logit += v * w.z;
      v = l.w + r.w; v = (v > 0.f) ? v : 0.2f * v; logit += v * w.w;
    }
    elog[(size_t)e * 4 + h] = logit;
    atomicMaxF(&smax[(size_t)d * 4 + h], logit);
  }
}

// Pass 2: a = exp(logit - max); denom += a; raw[dst] += a * gl[src]
__global__ void edge_accum(const int* __restrict__ eidx, int E, int nself,
                           const float* __restrict__ gl,
                           const float* __restrict__ elog,
                           const float* __restrict__ smax,
                           float* __restrict__ sden,
                           float* __restrict__ raw)
{
  const int e = blockIdx.x * blockDim.x + threadIdx.x;
  if (e >= E + nself) return;
  int s, d;
  if (e < E) { s = eidx[e]; d = eidx[E + e]; } else { s = e - E; d = s; }
  const float4* pl = (const float4*)(gl + (size_t)s * 64);
#pragma unroll
  for (int h = 0; h < 4; ++h) {
    const float a = __expf(elog[(size_t)e * 4 + h] - smax[(size_t)d * 4 + h]);
    atomicAdd(&sden[(size_t)d * 4 + h], a);
    float* pr = raw + (size_t)d * 64 + h * 16;
#pragma unroll
    for (int q = 0; q < 4; ++q) {
      const float4 l = pl[h * 4 + q];
      atomicAdd(&pr[q * 4 + 0], a * l.x);
      atomicAdd(&pr[q * 4 + 1], a * l.y);
      atomicAdd(&pr[q * 4 + 2], a * l.z);
      atomicAdd(&pr[q * 4 + 3], a * l.w);
    }
  }
}

// Pass 3: out[node,f] += raw/denom + bias[f]   (HeteroConv 'sum')
__global__ void seg_finish(const float* __restrict__ raw,
                           const float* __restrict__ sden,
                           const float* __restrict__ bias,
                           float* __restrict__ out, int nd)
{
  const int t = blockIdx.x * blockDim.x + threadIdx.x;
  if (t >= nd * 64) return;
  const int f = t % 64;
  const int node = t / 64;
  const float den = sden[(size_t)node * 4 + (f >> 4)];
  const float v = (den > 0.f) ? raw[t] / den : 0.f;
  out[t] += v + bias[f];
}

// r <- sigmoid(r) * h   (prepares candidate input [x | r*h])
__global__ void gate_rh(float* __restrict__ r, const float* __restrict__ h, int n)
{
  const int t = blockIdx.x * blockDim.x + threadIdx.x;
  if (t >= n) return;
  const float rg = 1.f / (1.f + __expf(-r[t]));
  r[t] = rg * h[t];
}

// h <- (1 - sigmoid(z)) * h + sigmoid(z) * tanh(t)
__global__ void gate_update(float* __restrict__ h, const float* __restrict__ z,
                            const float* __restrict__ tb, int n)
{
  const int t = blockIdx.x * blockDim.x + threadIdx.x;
  if (t >= n) return;
  const float zg = 1.f / (1.f + __expf(-z[t]));
  h[t] = (1.f - zg) * h[t] + zg * tanhf(tb[t]);
}

// copy first ncols of a [M,16]-padded matrix into tightly packed dst
__global__ void copy_cols(const float* __restrict__ src, float* __restrict__ dst,
                          int M, int ncols)
{
  const int t = blockIdx.x * blockDim.x + threadIdx.x;
  if (t >= M * ncols) return;
  const int m = t / ncols, c = t % ncols;
  dst[t] = src[(size_t)m * 16 + c];
}

// ---------------------------------------------------------------------------
// Host-side orchestration
// ---------------------------------------------------------------------------
struct SegBufs { float *gl, *gr, *elog, *smax, *sden, *raw; };

static inline dim3 g1(int n) { return dim3((n + 255) / 256); }
static inline dim3 gw(int M) { return dim3((M + 63) / 64); }   // 4 waves/block

static void run_gatv2(hipStream_t st,
                      const float* s0, const float* s1, int nsrc,
                      const float* d0, const float* d1, int ndst,
                      const int* eidx, int E, int nself,
                      const _Float16* Wl16, const float* bl,
                      const _Float16* Wr16, const float* br,
                      const float* att, const float* bias,
                      const SegBufs& sb, float* out)
{
  gemm_wmma<128, 64, 64><<<gw(nsrc), 128, 0, st>>>(
      s0, s1, Wl16, bl, 64, sb.gl, nsrc, 0);
  gemm_wmma<128, 64, 64><<<gw(ndst), 128, 0, st>>>(
      d0, d1, Wr16, br, 64, sb.gr, ndst, 0);
  const int Etot = E + nself;
  seg_init<<<g1(ndst * 64), 256, 0, st>>>(sb.raw, sb.smax, sb.sden, ndst);
  edge_logits_max<<<g1(Etot), 256, 0, st>>>(eidx, E, nself, sb.gl, sb.gr, att,
                                            sb.elog, sb.smax);
  edge_accum<<<g1(Etot), 256, 0, st>>>(eidx, E, nself, sb.gl, sb.elog,
                                       sb.smax, sb.sden, sb.raw);
  seg_finish<<<g1(ndst * 64), 256, 0, st>>>(sb.raw, sb.sden, bias, out, ndst);
}

extern "C" void kernel_launch(void* const* d_in, const int* in_sizes, int n_in,
                              void* d_out, int out_size, void* d_ws, size_t ws_size,
                              hipStream_t stream)
{
  // ---- input index map (top-level insertion order; `params` pytree flattened
  //      with sorted dict keys: cand, cell_dec, cell_enc, man_dec, man_enc,
  //      reset, update; each GAT dict as Wl, Wr, att, bias, bl, br;
  //      each conv dict as cc, cm, mc, mm) ----
  const float* F[100];
  for (int i = 0; i < 100 && i < n_in; ++i) F[i] = (const float*)d_in[i];
  const float* man_dyn   = F[0];
  const float* cell_dyn  = F[1];
  const float* man_stat  = F[2];
  const float* cell_stat = F[3];
  const int CAND = 4, CELL_DEC = 28, CELL_ENC = 32, MAN_DEC = 38, MAN_ENC = 42,
            RESET = 48, UPDATE = 72;
  const int* e_mm = (const int*)d_in[96];
  const int* e_cc = (const int*)d_in[97];
  const int* e_mc = (const int*)d_in[98];
  const int* e_cm = (const int*)d_in[99];
  float* out = (float*)d_out;

  // ---- workspace bump allocator ----
  char* ws = (char*)d_ws;
  size_t off = 0;
  auto alloc = [&](size_t bytes) -> void* {
    off = (off + 255) & ~(size_t)255;
    void* p = ws + off;
    off += bytes;
    return p;
  };

  // f16 weight pool (fragment-packed)
  _Float16* encm[3]; _Float16* encc[3];
  encm[0] = (_Float16*)alloc(32 * 64 * 2);
  encm[1] = (_Float16*)alloc(64 * 64 * 2);
  encm[2] = (_Float16*)alloc(64 * 64 * 2);
  encc[0] = (_Float16*)alloc(32 * 64 * 2);
  encc[1] = (_Float16*)alloc(64 * 64 * 2);
  encc[2] = (_Float16*)alloc(64 * 64 * 2);
  _Float16* decm0 = (_Float16*)alloc(64 * 64 * 2);
  _Float16* decm1 = (_Float16*)alloc(64 * 16 * 2);
  _Float16* decc0 = (_Float16*)alloc(64 * 64 * 2);
  _Float16* decc1 = (_Float16*)alloc(64 * 16 * 2);
  // conv projections: [gate 0..2 = reset,update,cand][type 0..3 = cc,cm,mc,mm][Wl,Wr]
  _Float16* convW[3][4][2];
  for (int g = 0; g < 3; ++g)
    for (int t = 0; t < 4; ++t)
      for (int w = 0; w < 2; ++w)
        convW[g][t][w] = (_Float16*)alloc(128 * 64 * 2);

  // fp32 buffers
  float* xm = (float*)alloc((size_t)kNM * 64 * 4);
  float* xc = (float*)alloc((size_t)kNC * 64 * 4);
  float* hm = (float*)alloc((size_t)kNM * 64 * 4);
  float* hc = (float*)alloc((size_t)kNC * 64 * 4);
  SegBufs sb;
  sb.gl   = (float*)alloc((size_t)kNC * 64 * 4);
  sb.gr   = (float*)alloc((size_t)kNC * 64 * 4);
  sb.elog = (float*)alloc((size_t)(kECC + kNC) * 4 * 4);
  sb.smax = (float*)alloc((size_t)kNC * 4 * 4);
  sb.sden = (float*)alloc((size_t)kNC * 4 * 4);
  sb.raw  = (float*)alloc((size_t)kNC * 64 * 4);
  float* rm = (float*)alloc((size_t)kNM * 64 * 4);   // reset gate / dec hidden
  float* rc = (float*)alloc((size_t)kNC * 64 * 4);
  float* zm = (float*)alloc((size_t)kNM * 64 * 4);   // update gate / dec out pad
  float* zc = (float*)alloc((size_t)kNC * 64 * 4);
  float* tm = (float*)alloc((size_t)kNM * 64 * 4);   // candidate
  float* tc = (float*)alloc((size_t)kNC * 64 * 4);
  float* inm = (float*)alloc((size_t)kNM * 32 * 4);
  float* inc = (float*)alloc((size_t)kNC * 32 * 4);
  (void)ws_size; (void)in_sizes; (void)out_size;

  // ---- pack weights into f16 B-fragments ----
  auto pack = [&](const float* src, _Float16* dst, int K, int N, int Kp, int Np) {
    pack_w16<<<g1((Kp / 32) * (Np / 16) * 512), 256, 0, stream>>>(src, dst, K, N, Kp, Np);
  };
  pack(F[MAN_ENC + 0], encm[0], 6, 64, 32, 64);
  pack(F[MAN_ENC + 2], encm[1], 64, 64, 64, 64);
  pack(F[MAN_ENC + 4], encm[2], 64, 64, 64, 64);
  pack(F[CELL_ENC + 0], encc[0], 9, 64, 32, 64);
  pack(F[CELL_ENC + 2], encc[1], 64, 64, 64, 64);
  pack(F[CELL_ENC + 4], encc[2], 64, 64, 64, 64);
  pack(F[MAN_DEC + 0], decm0, 64, 64, 64, 64);
  pack(F[MAN_DEC + 2], decm1, 64, 2, 64, 16);
  pack(F[CELL_DEC + 0], decc0, 64, 64, 64, 64);
  pack(F[CELL_DEC + 2], decc1, 64, 3, 64, 16);
  const int gateBase[3] = {RESET, UPDATE, CAND};
  for (int g = 0; g < 3; ++g)
    for (int t = 0; t < 4; ++t) {
      pack(F[gateBase[g] + t * 6 + 0], convW[g][t][0], 128, 64, 128, 64);
      pack(F[gateBase[g] + t * 6 + 1], convW[g][t][1], 128, 64, 128, 64);
    }

  // ---- encoders ----
  const float dscale = 1.0f / (1.0f + 1e-6f);
  pack_inputs<<<g1(kNM * 32), 256, 0, stream>>>(man_stat, 4, man_dyn, 2, inm, kNM, dscale);
  pack_inputs<<<g1(kNC * 32), 256, 0, stream>>>(cell_stat, 6, cell_dyn, 3, inc, kNC, dscale);
  gemm_wmma<32, 32, 64><<<gw(kNM), 128, 0, stream>>>(inm, inm, encm[0], F[MAN_ENC + 1], 64, sb.gl, kNM, 1);
  gemm_wmma<64, 64, 64><<<gw(kNM), 128, 0, stream>>>(sb.gl, sb.gl, encm[1], F[MAN_ENC + 3], 64, sb.gr, kNM, 1);
  gemm_wmma<64, 64, 64><<<gw(kNM), 128, 0, stream>>>(sb.gr, sb.gr, encm[2], F[MAN_ENC + 5], 64, xm, kNM, 0);
  gemm_wmma<32, 32, 64><<<gw(kNC), 128, 0, stream>>>(inc, inc, encc[0], F[CELL_ENC + 1], 64, sb.gl, kNC, 1);
  gemm_wmma<64, 64, 64><<<gw(kNC), 128, 0, stream>>>(sb.gl, sb.gl, encc[1], F[CELL_ENC + 3], 64, sb.gr, kNC, 1);
  gemm_wmma<64, 64, 64><<<gw(kNC), 128, 0, stream>>>(sb.gr, sb.gr, encc[2], F[CELL_ENC + 5], 64, xc, kNC, 0);
  fill_f32<<<g1(kNM * 64), 256, 0, stream>>>(hm, kNM * 64, 0.f);
  fill_f32<<<g1(kNC * 64), 256, 0, stream>>>(hc, kNC * 64, 0.f);

  // ---- one HeteroConv gate: out_m/out_c = sum over 4 GATv2 edge types ----
  auto hetero = [&](int g, int pb, const float* am, const float* ac,
                    float* om, float* oc) {
    fill_f32<<<g1(kNM * 64), 256, 0, stream>>>(om, kNM * 64, 0.f);
    fill_f32<<<g1(kNC * 64), 256, 0, stream>>>(oc, kNC * 64, 0.f);
    // cc : cell -> cell (with self loops)
    run_gatv2(stream, xc, ac, kNC, xc, ac, kNC, e_cc, kECC, kNC,
              convW[g][0][0], F[pb + 0 * 6 + 4], convW[g][0][1], F[pb + 0 * 6 + 5],
              F[pb + 0 * 6 + 2], F[pb + 0 * 6 + 3], sb, oc);
    // cm : cell -> manhole
    run_gatv2(stream, xc, ac, kNC, xm, am, kNM, e_cm, kECM, 0,
              convW[g][1][0], F[pb + 1 * 6 + 4], convW[g][1][1], F[pb + 1 * 6 + 5],
              F[pb + 1 * 6 + 2], F[pb + 1 * 6 + 3], sb, om);
    // mc : manhole -> cell
    run_gatv2(stream, xm, am, kNM, xc, ac, kNC, e_mc, kEMC, 0,
              convW[g][2][0], F[pb + 2 * 6 + 4], convW[g][2][1], F[pb + 2 * 6 + 5],
              F[pb + 2 * 6 + 2], F[pb + 2 * 6 + 3], sb, oc);
    // mm : manhole -> manhole (with self loops)
    run_gatv2(stream, xm, am, kNM, xm, am, kNM, e_mm, kEMM, kNM,
              convW[g][3][0], F[pb + 3 * 6 + 4], convW[g][3][1], F[pb + 3 * 6 + 5],
              F[pb + 3 * 6 + 2], F[pb + 3 * 6 + 3], sb, om);
  };

  // ---- GRU recurrent steps ----
  for (int step = 0; step < 2; ++step) {
    hetero(0, RESET,  hm, hc, rm, rc);                 // r pre-activation
    hetero(1, UPDATE, hm, hc, zm, zc);                 // z pre-activation
    gate_rh<<<g1(kNM * 64), 256, 0, stream>>>(rm, hm, kNM * 64);  // r <- sig(r)*h
    gate_rh<<<g1(kNC * 64), 256, 0, stream>>>(rc, hc, kNC * 64);
    hetero(2, CAND, rm, rc, tm, tc);                   // candidate pre-activation
    gate_update<<<g1(kNM * 64), 256, 0, stream>>>(hm, zm, tm, kNM * 64);
    gate_update<<<g1(kNC * 64), 256, 0, stream>>>(hc, zc, tc, kNC * 64);
  }

  // ---- decoders ----
  gemm_wmma<64, 64, 64><<<gw(kNM), 128, 0, stream>>>(hm, hm, decm0, F[MAN_DEC + 1], 64, rm, kNM, 1);
  gemm_wmma<64, 64, 16><<<gw(kNM), 128, 0, stream>>>(rm, rm, decm1, F[MAN_DEC + 3], 2, zm, kNM, 0);
  gemm_wmma<64, 64, 64><<<gw(kNC), 128, 0, stream>>>(hc, hc, decc0, F[CELL_DEC + 1], 64, rc, kNC, 1);
  gemm_wmma<64, 64, 16><<<gw(kNC), 128, 0, stream>>>(rc, rc, decc1, F[CELL_DEC + 3], 3, zc, kNC, 0);

  // ---- pack outputs: pred_m, pred_c, h_m, h_c ----
  copy_cols<<<g1(kNM * 2), 256, 0, stream>>>(zm, out, kNM, 2);
  copy_cols<<<g1(kNC * 3), 256, 0, stream>>>(zc, out + (size_t)kNM * 2, kNC, 3);
  hipMemcpyAsync(out + (size_t)kNM * 2 + (size_t)kNC * 3, hm,
                 (size_t)kNM * 64 * sizeof(float), hipMemcpyDeviceToDevice, stream);
  hipMemcpyAsync(out + (size_t)kNM * 2 + (size_t)kNC * 3 + (size_t)kNM * 64, hc,
                 (size_t)kNC * 64 * sizeof(float), hipMemcpyDeviceToDevice, stream);
}
